// UserNet_52321291600459
// MI455X (gfx1250) — compile-verified, matrix operand order checked by reading, not run on hardware
//
#include <hip/hip_runtime.h>

typedef __attribute__((ext_vector_type(16))) _Float16 v16h;
typedef __attribute__((ext_vector_type(8)))  float    v8f;

namespace {

constexpr int kB  = 4096;
constexpr int kL  = 200;
constexpr int kE  = 128;
constexpr int kF  = 64;
constexpr int kLP = 208;            // rows padded to 13 tiles of 16
constexpr int K1  = kE + kF;        // 192
constexpr int K2  = kE;             // 128
constexpr int K3  = 3 * kE;         // 384

// ---- LDS layout (bytes) ----
constexpr int OFF_ITEMS = 0;                        // f32 [208][128]
constexpr int SZ_ITEMS  = kLP * kE * 4;             // 106496
constexpr int OFF_FX    = OFF_ITEMS + SZ_ITEMS;     // f16 [208][128]
constexpr int SZ_HROWS  = kLP * kE * 2;             // 53248
constexpr int OFF_COMP  = OFF_FX + SZ_HROWS;        // f16 [208][128]
constexpr int OFF_W     = OFF_COMP + SZ_HROWS;      // f16 weight staging
constexpr int SZ_W      = kE * K3 * 2;              // 98304 (largest: w1)
constexpr int OFF_FEAT  = OFF_W + kE * K1 * 2;      // f16 [208][64] (phase1 only)
constexpr int OFF_USER  = OFF_W + SZ_W;             // f32 [128]
constexpr int OFF_USERH = OFF_USER + kE * 4;        // f16 [128]
constexpr int OFF_FB1   = OFF_USERH + kE * 2;       // f32 [128]
constexpr int OFF_FB2   = OFF_FB1 + kE * 4;         // f32 [128]
constexpr int OFF_B1    = OFF_FB2 + kE * 4;         // f32 [128]
constexpr int OFF_W2    = OFF_B1 + kE * 4;          // f32 [128]
constexpr int OFF_LOG   = OFF_W2 + kE * 4;          // f32 [208] (pad entries unused)
constexpr int OFF_SCAL  = OFF_LOG + kLP * 4;        // f32 [2]
constexpr int OFF_PART  = OFF_SCAL + 16;            // f32 [4][128] partial sums
constexpr int SMEM_BYTES = OFF_PART + 4 * kE * 4;   // ~317 KB < 320 KB

union FragH { v16h h; unsigned int u[8]; };

// K offset inside a 32-wide K tile for packed-pair v of the fragment.
// 16-bit A/B layout: VGPR v<4 holds K = 2v,2v+1 (+8 for lanes 16..31),
// VGPR v>=4 holds K = 16+2(v-4) (+8 for upper lanes).
__device__ __forceinline__ int kmap(int v, int half) {
    return ((v & 3) * 2) + ((v >> 2) * 16) + half * 8;
}

// Load a 16x32 f16 A/B fragment from an f16 row in LDS (row-major, contiguous K).
// Pair offsets within each K-half are contiguous -> backend emits 2x ds_load_b128.
__device__ __forceinline__ v16h load_frag_h(const _Float16* row, int kbase, int half) {
    FragH f;
#pragma unroll
    for (int v = 0; v < 8; ++v) {
        int k = kbase + kmap(v, half);                 // even -> dword aligned
        f.u[v] = *(const unsigned int*)(row + k);
    }
    return f.h;
}

// Same, but source row is f32 in LDS; convert to f16 pairs on the fly.
__device__ __forceinline__ v16h load_frag_f32(const float* row, int kbase, int half) {
    FragH f;
#pragma unroll
    for (int v = 0; v < 8; ++v) {
        int k = kbase + kmap(v, half);
        union { _Float16 h[2]; unsigned int u; } p;
        p.h[0] = (_Float16)row[k];
        p.h[1] = (_Float16)row[k + 1];
        f.u[v] = p.u;
    }
    return f.h;
}

// Convert 4 f32 -> 4 f16 and store as one 8-byte LDS write.
__device__ __forceinline__ void cvt_store4(_Float16* dst, float4 v) {
    union { _Float16 h[4]; uint2 u; } t;
    t.h[0] = (_Float16)v.x; t.h[1] = (_Float16)v.y;
    t.h[2] = (_Float16)v.z; t.h[3] = (_Float16)v.w;
    *(uint2*)dst = t.u;
}

} // namespace

__launch_bounds__(512, 1)
__global__ void usernet_fused_kernel(const int* __restrict__ user_ids,
                                     const int* __restrict__ item_ids,
                                     const float* __restrict__ features,
                                     const float* __restrict__ user_emb,
                                     const float* __restrict__ item_emb,
                                     const float* __restrict__ feat_w1,
                                     const float* __restrict__ feat_b1,
                                     const float* __restrict__ feat_w2,
                                     const float* __restrict__ feat_b2,
                                     const float* __restrict__ w1,
                                     const float* __restrict__ b1,
                                     const float* __restrict__ w2,
                                     float* __restrict__ out) {
    extern __shared__ __align__(16) unsigned char smem[];
    float*    s_items = (float*)   (smem + OFF_ITEMS);
    _Float16* s_fx    = (_Float16*)(smem + OFF_FX);
    _Float16* s_comp  = (_Float16*)(smem + OFF_COMP);
    _Float16* s_wh    = (_Float16*)(smem + OFF_W);
    _Float16* s_feat  = (_Float16*)(smem + OFF_FEAT);
    float*    s_user  = (float*)   (smem + OFF_USER);
    _Float16* s_userh = (_Float16*)(smem + OFF_USERH);
    float*    s_fb1   = (float*)   (smem + OFF_FB1);
    float*    s_fb2   = (float*)   (smem + OFF_FB2);
    float*    s_b1    = (float*)   (smem + OFF_B1);
    float*    s_w2    = (float*)   (smem + OFF_W2);
    float*    s_log   = (float*)   (smem + OFF_LOG);
    float*    s_scal  = (float*)   (smem + OFF_SCAL);
    float*    s_part  = (float*)   (smem + OFF_PART);

    const int tid  = threadIdx.x;
    const int nthr = blockDim.x;           // 512
    const int bidx = blockIdx.x;
    const int wave = tid >> 5;             // 0..15
    const int lane = tid & 31;
    const int half = lane >> 4;            // 0/1: K-half selector
    const int nlo  = lane & 15;            // column within N tile
    const int nt   = wave & 7;             // N tile (8 tiles of 16 -> E=128)
    const int mst  = wave >> 3;            // M stripe start (mt = mst, mst+2, ...)
    const int e_col = nt * 16 + nlo;       // this lane's output channel

    // ---------------- Phase 0: stage everything into LDS ----------------
    const int uid = user_ids[bidx];
    for (int i = tid; i < kE; i += nthr) {
        float v = user_emb[uid * kE + i];
        s_user[i]  = v;
        s_userh[i] = (_Float16)v;
        s_fb1[i] = feat_b1[i];
        s_fb2[i] = feat_b2[i];
        s_b1[i]  = b1[i];
        s_w2[i]  = w2[i];
    }
    // items gather (f32, float4 vectorized); pad rows zeroed
    {
        const float4* src4 = (const float4*)item_emb;
        float4* dst4 = (float4*)s_items;
        for (int i = tid; i < kLP * (kE / 4); i += nthr) {
            int row = i >> 5, c = i & 31;
            float4 v = make_float4(0.f, 0.f, 0.f, 0.f);
            if (row < kL) {
                int it = item_ids[bidx * kL + row];
                v = src4[it * (kE / 4) + c];
            }
            dst4[i] = v;
        }
    }
    // features -> f16 (pad rows zero)
    {
        const float4* src4 = (const float4*)features;
        for (int i = tid; i < kLP * (kF / 4); i += nthr) {
            int row = i >> 4, c = i & 15;
            float4 v = make_float4(0.f, 0.f, 0.f, 0.f);
            if (row < kL) v = src4[(bidx * kL + row) * (kF / 4) + c];
            cvt_store4(s_feat + row * kF + c * 4, v);
        }
    }
    // feat_w1 -> f16 staging
    {
        const float4* src4 = (const float4*)feat_w1;
        for (int i = tid; i < (kE * K1) / 4; i += nthr)
            cvt_store4(s_wh + i * 4, src4[i]);
    }
    for (int i = tid; i < kLP; i += nthr) s_log[i] = 0.f;
    __syncthreads();

    // ---------------- Phase 1: fx = relu([user|feat] @ feat_w1^T + fb1) ----------------
    {
        const _Float16* wrow = s_wh + e_col * K1;
        const float bia = s_fb1[e_col];
        for (int mt = mst; mt < kLP / 16; mt += 2) {
            const int lrow = mt * 16 + nlo;      // A-matrix row for this lane
            v8f acc = {};
#pragma unroll
            for (int kt = 0; kt < K1 / 32; ++kt) {
                const int kb = kt * 32;
                v16h a = (kt < 4) ? load_frag_h(s_userh, kb, half)
                                  : load_frag_h(s_feat + lrow * kF, kb - kE, half);
                v16h bf = load_frag_h(wrow, kb, half);
                acc = __builtin_amdgcn_wmma_f32_16x16x32_f16(
                        false, a, false, bf, (short)0, acc, false, false);
            }
            // Unguarded: pad rows (200..207) are in-bounds scratch, never observed.
#pragma unroll
            for (int r = 0; r < 8; ++r) {
                int l = mt * 16 + r + half * 8;
                s_fx[l * kE + e_col] = (_Float16)fmaxf(acc[r] + bia, 0.f);
            }
        }
    }
    __syncthreads();

    // stage feat_w2 (overwrites fw1 region)
    {
        const float4* src4 = (const float4*)feat_w2;
        for (int i = tid; i < (kE * K2) / 4; i += nthr)
            cvt_store4(s_wh + i * 4, src4[i]);
    }
    __syncthreads();

    // ---------------- Phase 2: comp = relu(fx @ feat_w2^T + fb2) ----------------
    {
        const _Float16* wrow = s_wh + e_col * K2;
        const float bia = s_fb2[e_col];
        for (int mt = mst; mt < kLP / 16; mt += 2) {
            const int lrow = mt * 16 + nlo;
            v8f acc = {};
#pragma unroll
            for (int kt = 0; kt < K2 / 32; ++kt) {
                const int kb = kt * 32;
                v16h a  = load_frag_h(s_fx + lrow * kE, kb, half);
                v16h bf = load_frag_h(wrow, kb, half);
                acc = __builtin_amdgcn_wmma_f32_16x16x32_f16(
                        false, a, false, bf, (short)0, acc, false, false);
            }
#pragma unroll
            for (int r = 0; r < 8; ++r) {
                int l = mt * 16 + r + half * 8;
                s_comp[l * kE + e_col] = (_Float16)fmaxf(acc[r] + bia, 0.f);
            }
        }
    }
    __syncthreads();

    // stage w1 (full 98 KB region)
    {
        const float4* src4 = (const float4*)w1;
        for (int i = tid; i < (kE * K3) / 4; i += nthr)
            cvt_store4(s_wh + i * 4, src4[i]);
    }
    __syncthreads();

    // -------- Phase 3: h = relu([user|item|comp] @ w1^T + b1); logit += h.w2 --------
    {
        const _Float16* wrow = s_wh + e_col * K3;
        const float bia = s_b1[e_col];
        const float w2v = s_w2[e_col];
        for (int mt = mst; mt < kLP / 16; mt += 2) {
            const int lrow = mt * 16 + nlo;
            v8f acc = {};
#pragma unroll
            for (int kt = 0; kt < K3 / 32; ++kt) {
                const int kb = kt * 32;
                v16h a;
                if (kt < 4)      a = load_frag_h(s_userh, kb, half);
                else if (kt < 8) a = load_frag_f32(s_items + lrow * kE, kb - kE, half);
                else             a = load_frag_h(s_comp + lrow * kE, kb - 2 * kE, half);
                v16h bf = load_frag_h(wrow, kb, half);
                acc = __builtin_amdgcn_wmma_f32_16x16x32_f16(
                        false, a, false, bf, (short)0, acc, false, false);
            }
            float part[8];
#pragma unroll
            for (int r = 0; r < 8; ++r)
                part[r] = fmaxf(acc[r] + bia, 0.f) * w2v;
            // reduce over the 16 N-lanes of each half-wave
#pragma unroll
            for (int r = 0; r < 8; ++r) {
                part[r] += __shfl_xor(part[r], 1);
                part[r] += __shfl_xor(part[r], 2);
                part[r] += __shfl_xor(part[r], 4);
                part[r] += __shfl_xor(part[r], 8);
            }
            if (nlo == 0) {
#pragma unroll
                for (int r = 0; r < 8; ++r) {
                    int l = mt * 16 + r + half * 8;
                    atomicAdd(&s_log[l], part[r]);   // pad entries are dead scratch
                }
            }
        }
    }
    __syncthreads();

    // ---------------- Softmax over L (b2 cancels: shift invariance) ----------------
    if (wave == 0) {
        float m = -INFINITY;
        for (int l = lane; l < kL; l += 32) m = fmaxf(m, s_log[l]);
#pragma unroll
        for (int s = 16; s > 0; s >>= 1) m = fmaxf(m, __shfl_xor(m, s));
        if (lane == 0) s_scal[0] = m;
    }
    __syncthreads();
    {
        float mx = s_scal[0];
        for (int l = tid; l < kL; l += nthr) s_log[l] = __expf(s_log[l] - mx);
    }
    __syncthreads();
    if (wave == 0) {
        float s = 0.f;
        for (int l = lane; l < kL; l += 32) s += s_log[l];
#pragma unroll
        for (int sh = 16; sh > 0; sh >>= 1) s += __shfl_xor(s, sh);
        if (lane == 0) s_scal[1] = s;
    }
    __syncthreads();

    // ---------------- out = user + sum_l items[l] * attn[l] ----------------
    {
        const int g = tid >> 7;          // 0..3 -> L chunk of 50
        const int e = tid & 127;
        const int l0 = g * 50;
        float acc = 0.f;
        for (int l = l0; l < l0 + 50; ++l) acc += s_items[l * kE + e] * s_log[l];
        s_part[g * kE + e] = acc;
    }
    __syncthreads();
    if (tid < kE) {
        const float inv = 1.0f / s_scal[1];
        float r = s_part[tid] + s_part[kE + tid] + s_part[2 * kE + tid] + s_part[3 * kE + tid];
        out[bidx * kE + tid] = s_user[tid] + r * inv;
    }
}

extern "C" void kernel_launch(void* const* d_in, const int* in_sizes, int n_in,
                              void* d_out, int out_size, void* d_ws, size_t ws_size,
                              hipStream_t stream) {
    (void)in_sizes; (void)n_in; (void)out_size; (void)d_ws; (void)ws_size;
    const int*   user_ids = (const int*)  d_in[0];
    const int*   item_ids = (const int*)  d_in[1];
    const float* features = (const float*)d_in[2];
    const float* user_emb = (const float*)d_in[3];
    const float* item_emb = (const float*)d_in[4];
    const float* feat_w1  = (const float*)d_in[5];
    const float* feat_b1  = (const float*)d_in[6];
    const float* feat_w2  = (const float*)d_in[7];
    const float* feat_b2  = (const float*)d_in[8];
    const float* w1       = (const float*)d_in[9];
    const float* b1       = (const float*)d_in[10];
    const float* w2       = (const float*)d_in[11];
    // d_in[12] = b2: scalar shift inside softmax -> mathematically a no-op.
    float* out = (float*)d_out;

    hipFuncSetAttribute(reinterpret_cast<const void*>(usernet_fused_kernel),
                        hipFuncAttributeMaxDynamicSharedMemorySize, SMEM_BYTES);

    usernet_fused_kernel<<<kB, 512, SMEM_BYTES, stream>>>(
        user_ids, item_ids, features, user_emb, item_emb,
        feat_w1, feat_b1, feat_w2, feat_b2, w1, b1, w2, out);
}
